// GraphTransformer_71829033058359
// MI455X (gfx1250) — compile-verified, matrix operand order checked by reading, not run on hardware
//
#include <hip/hip_runtime.h>
#include <hip/hip_bf16.h>

// Problem constants (match reference)
#define N_NODES 50000
#define N_EDGES 800000
#define K_DIM   128     // feature width into each GEMM
#define D_CAT   512     // q|k|v|s concatenated output width
#define BSTRIDE 136     // padded LDS row stride (ushorts): 272B breaks 256B bank period

typedef __attribute__((ext_vector_type(16))) __bf16 v16bf;
typedef __attribute__((ext_vector_type(8)))  float  v8f;

struct Frag32B { uint4 lo; uint4 hi; };

__device__ __forceinline__ unsigned short f2bf(float f) {
  unsigned int u = __float_as_uint(f);
  unsigned int r = u + 0x7FFFu + ((u >> 16) & 1u);   // round-to-nearest-even
  return (unsigned short)(r >> 16);
}

// Monotone float -> uint encoding so atomicMax(u32) implements float segment-max.
__device__ __forceinline__ unsigned int fenc(float f) {
  unsigned int u = __float_as_uint(f);
  return (u & 0x80000000u) ? ~u : (u | 0x80000000u);
}
__device__ __forceinline__ float fdec(unsigned int k) {
  unsigned int u = (k & 0x80000000u) ? (k ^ 0x80000000u) : ~k;
  return __uint_as_float(u);
}
#define ENC_NEG_INF 0x007FFFFFu   // fenc(-inf)

// ---------------------------------------------------------------- utilities
__global__ void k_fill_u32(unsigned int* __restrict__ p, unsigned int v, int n) {
  int i = blockIdx.x * blockDim.x + threadIdx.x;
  if (i < n) p[i] = v;
}

__global__ void k_cvt_bf16(const float* __restrict__ x,
                           unsigned short* __restrict__ y, int n) {
  int i = blockIdx.x * blockDim.x + threadIdx.x;
  if (i < n) y[i] = f2bf(x[i]);
}

// Pack 4 (128x128) fp32 weights into out-col-major bf16 (512 x 128) + concat bias.
__global__ void k_pack_w(const float* __restrict__ w0, const float* __restrict__ w1,
                         const float* __restrict__ w2, const float* __restrict__ w3,
                         const float* __restrict__ b0, const float* __restrict__ b1,
                         const float* __restrict__ b2, const float* __restrict__ b3,
                         unsigned short* __restrict__ wt, float* __restrict__ bias) {
  int idx = blockIdx.x * blockDim.x + threadIdx.x;      // 512*128 threads
  if (idx >= D_CAT * K_DIM) return;
  int col = idx >> 7;          // output column 0..511
  int kk  = idx & 127;         // reduction index 0..127
  int g = col >> 7, n = col & 127;
  const float* w = (g == 0) ? w0 : (g == 1) ? w1 : (g == 2) ? w2 : w3;
  wt[(size_t)col * K_DIM + kk] = f2bf(w[(size_t)kk * 128 + n]);
  if (kk == 0) {
    const float* b = (g == 0) ? b0 : (g == 1) ? b1 : (g == 2) ? b2 : b3;
    bias[col] = b[n];
  }
}

// ---------------------------------------------------------------- WMMA GEMM
// C[M x 512] = A[M x 128](bf16) * W^T(512 x 128 bf16, row = out col) + bias
// Block = 256 threads = 8 waves; block tile 128(M) x 128(cols); wave = 16 x 128.
// Weight tile (32 KB, shared by all 8 waves) is staged in LDS with the CDNA5
// async global->LDS path (ASYNCcnt), padded to avoid bank conflicts.
__global__ void __launch_bounds__(256)
k_gemm_bf16(const unsigned short* __restrict__ A,
            const unsigned short* __restrict__ W,
            const float* __restrict__ bias,
            float* __restrict__ C, int M) {
  __shared__ unsigned short Bs[128 * BSTRIDE];   // ~34 KB

  const int tid  = threadIdx.x;
  const int wave = tid >> 5;
  const int lane = tid & 31;
  const int l15  = lane & 15;
  const int kb   = (lane >> 4) << 3;       // 0 | 8 : K-subblock owned by this half-wave
  const int rowBase = blockIdx.x * 128 + wave * 16;
  const int colBase = blockIdx.y * 128;

  // ---- async copy: 128 weight rows (cols of the GEMM) x 256B -> padded LDS rows.
  {
    const unsigned short* gsrc = W + (size_t)colBase * K_DIM;   // contiguous 32KB block
#pragma unroll
    for (int i = 0; i < 8; ++i) {
      int chunk = tid + i * 256;              // 2048 x 16B chunks
      int row = chunk >> 4, seg = chunk & 15;
      unsigned lds_off = (unsigned)(size_t)(&Bs[row * BSTRIDE + seg * 8]);
      unsigned long long gaddr =
          (unsigned long long)(size_t)(gsrc + (size_t)row * K_DIM + seg * 8);
      asm volatile("global_load_async_to_lds_b128 %0, %1, off"
                   :: "v"(lds_off), "v"(gaddr) : "memory");
    }
  }

  // ---- preload this wave's A fragments for all four K-steps (overlaps the async DMA)
  int arow = rowBase + l15;
  if (arow > M - 1) arow = M - 1;            // clamp ragged tile (stores are guarded)
  const unsigned short* ap = A + (size_t)arow * K_DIM + kb;
  Frag32B fa[4];
#pragma unroll
  for (int k = 0; k < 4; ++k) {
    fa[k].lo = *(const uint4*)(ap + k * 32);        // K = 32k+kb   .. +7
    fa[k].hi = *(const uint4*)(ap + k * 32 + 16);   // K = 32k+16+kb .. +7
  }

  asm volatile("s_wait_asynccnt 0x0" ::: "memory");
  __syncthreads();

  v8f acc[8];
#pragma unroll
  for (int t = 0; t < 8; ++t) acc[t] = (v8f)0.0f;

#pragma unroll
  for (int ks = 0; ks < 4; ++ks) {
    v16bf a = __builtin_bit_cast(v16bf, fa[ks]);
#pragma unroll
    for (int t = 0; t < 8; ++t) {
      const unsigned short* bp = &Bs[(t * 16 + l15) * BSTRIDE + ks * 32 + kb];
      Frag32B fb;
      fb.lo = *(const uint4*)bp;
      fb.hi = *(const uint4*)(bp + 16);
      v16bf b = __builtin_bit_cast(v16bf, fb);
      acc[t] = __builtin_amdgcn_wmma_f32_16x16x32_bf16(
                   false, a, false, b, (short)0, acc[t], false, false);
    }
  }

  const int rhalf = (lane >> 4) << 3;   // C/D layout: lanes 16-31 hold rows M+8
#pragma unroll
  for (int t = 0; t < 8; ++t) {
    int col = colBase + t * 16 + l15;
    float bc = bias[col];
#pragma unroll
    for (int r = 0; r < 8; ++r) {
      int row = rowBase + r + rhalf;
      if (row < M) C[(size_t)row * D_CAT + col] = acc[t][r] + bc;
    }
  }
}

// ---------------------------------------------------------------- edge passes
// One wave per edge. C holds q|k|v|s at offsets 0/128/256/384, row stride 512.

// Layer-1 logits (heads=4, ch=32): 8-lane dot reduction per head + segment max.
__global__ void __launch_bounds__(256)
k_edge_logits4(const int* __restrict__ ei, const float* __restrict__ C,
               float* __restrict__ logits, unsigned int* __restrict__ nmax, int E) {
  int wid  = blockIdx.x * 8 + (threadIdx.x >> 5);
  int lane = threadIdx.x & 31;
  if (wid >= E) return;
  int src = ei[wid];
  int dst = ei[E + wid];
  const float4* q = (const float4*)(C + (size_t)dst * D_CAT);
  const float4* k = (const float4*)(C + (size_t)src * D_CAT + 128);
  float4 a = q[lane], b = k[lane];
  float d = a.x * b.x + a.y * b.y + a.z * b.z + a.w * b.w;
  d += __shfl_xor(d, 1, 32);
  d += __shfl_xor(d, 2, 32);
  d += __shfl_xor(d, 4, 32);                       // 8 lanes = one head (ch 32)
  if ((lane & 7) == 0) {
    int h = lane >> 3;
    float lg = d * 0.17677669529663687f;           // 1/sqrt(32)
    logits[(size_t)wid * 4 + h] = lg;
    atomicMax(&nmax[(size_t)dst * 4 + h], fenc(lg));
  }
}

// Layer-1 fused exp/sum/weighted-message (heads=4).
__global__ void __launch_bounds__(256)
k_edge_agg4(const int* __restrict__ ei, const float* __restrict__ C,
            const float* __restrict__ logits, const unsigned int* __restrict__ nmax,
            float* __restrict__ nsum, float* __restrict__ accum, int E) {
  int wid  = blockIdx.x * 8 + (threadIdx.x >> 5);
  int lane = threadIdx.x & 31;
  if (wid >= E) return;
  int src = ei[wid];
  int dst = ei[E + wid];
  int h = lane >> 3;
  float m  = fdec(nmax[(size_t)dst * 4 + h]);
  float e  = __expf(logits[(size_t)wid * 4 + h] - m);
  if ((lane & 7) == 0) unsafeAtomicAdd(&nsum[(size_t)dst * 4 + h], e);
  const float4* v = (const float4*)(C + (size_t)src * D_CAT + 256);
  float4 vv = v[lane];
  float* ac = accum + (size_t)dst * 128 + lane * 4;
  unsafeAtomicAdd(ac + 0, e * vv.x);
  unsafeAtomicAdd(ac + 1, e * vv.y);
  unsafeAtomicAdd(ac + 2, e * vv.z);
  unsafeAtomicAdd(ac + 3, e * vv.w);
}

// Layer-2 logits (heads=1, ch=128): full-wave reduction.
__global__ void __launch_bounds__(256)
k_edge_logits1(const int* __restrict__ ei, const float* __restrict__ C,
               float* __restrict__ logits, unsigned int* __restrict__ nmax, int E) {
  int wid  = blockIdx.x * 8 + (threadIdx.x >> 5);
  int lane = threadIdx.x & 31;
  if (wid >= E) return;
  int src = ei[wid];
  int dst = ei[E + wid];
  const float4* q = (const float4*)(C + (size_t)dst * D_CAT);
  const float4* k = (const float4*)(C + (size_t)src * D_CAT + 128);
  float4 a = q[lane], b = k[lane];
  float d = a.x * b.x + a.y * b.y + a.z * b.z + a.w * b.w;
  d += __shfl_xor(d, 1, 32);
  d += __shfl_xor(d, 2, 32);
  d += __shfl_xor(d, 4, 32);
  d += __shfl_xor(d, 8, 32);
  d += __shfl_xor(d, 16, 32);
  if (lane == 0) {
    float lg = d * 0.08838834764831845f;           // 1/sqrt(128)
    logits[wid] = lg;
    atomicMax(&nmax[dst], fenc(lg));
  }
}

__global__ void __launch_bounds__(256)
k_edge_agg1(const int* __restrict__ ei, const float* __restrict__ C,
            const float* __restrict__ logits, const unsigned int* __restrict__ nmax,
            float* __restrict__ nsum, float* __restrict__ accum, int E) {
  int wid  = blockIdx.x * 8 + (threadIdx.x >> 5);
  int lane = threadIdx.x & 31;
  if (wid >= E) return;
  int src = ei[wid];
  int dst = ei[E + wid];
  float m = fdec(nmax[dst]);
  float e = __expf(logits[wid] - m);
  if (lane == 0) unsafeAtomicAdd(&nsum[dst], e);
  const float4* v = (const float4*)(C + (size_t)src * D_CAT + 256);
  float4 vv = v[lane];
  float* ac = accum + (size_t)dst * 128 + lane * 4;
  unsafeAtomicAdd(ac + 0, e * vv.x);
  unsafeAtomicAdd(ac + 1, e * vv.y);
  unsafeAtomicAdd(ac + 2, e * vv.z);
  unsafeAtomicAdd(ac + 3, e * vv.w);
}

// ---------------------------------------------------------------- finalize
// Layer 1: h = relu(accum/(sum+eps) + skip), emit bf16 for the next GEMM.
__global__ void k_final_l1(const float* __restrict__ accum, const float* __restrict__ nsum,
                           const float* __restrict__ C, unsigned short* __restrict__ hbf) {
  int idx = blockIdx.x * blockDim.x + threadIdx.x;
  if (idx >= N_NODES * 128) return;
  int n = idx >> 7, c = idx & 127, h = c >> 5;
  float s = nsum[(size_t)n * 4 + h];
  float o = accum[idx] / (s + 1e-16f) + C[(size_t)n * D_CAT + 384 + c];
  hbf[idx] = f2bf(fmaxf(o, 0.0f));
}

// Layer 2: out = accum/(sum+eps) + skip  (fp32 result).
__global__ void k_final_l2(const float* __restrict__ accum, const float* __restrict__ nsum,
                           const float* __restrict__ C, float* __restrict__ out) {
  int idx = blockIdx.x * blockDim.x + threadIdx.x;
  if (idx >= N_NODES * 128) return;
  int n = idx >> 7, c = idx & 127;
  out[idx] = accum[idx] / (nsum[n] + 1e-16f) + C[(size_t)n * D_CAT + 384 + c];
}

// ---------------------------------------------------------------- launcher
extern "C" void kernel_launch(void* const* d_in, const int* in_sizes, int n_in,
                              void* d_out, int out_size, void* d_ws, size_t ws_size,
                              hipStream_t stream) {
  (void)in_sizes; (void)n_in; (void)out_size; (void)ws_size;
  const float* x  = (const float*)d_in[0];
  const int*   ei = (const int*)d_in[1];            // (2,E): row0=src, row1=dst
  const float *w1q = (const float*)d_in[2],  *b1q = (const float*)d_in[3];
  const float *w1k = (const float*)d_in[4],  *b1k = (const float*)d_in[5];
  const float *w1v = (const float*)d_in[6],  *b1v = (const float*)d_in[7];
  const float *w1s = (const float*)d_in[8],  *b1s = (const float*)d_in[9];
  const float *w2q = (const float*)d_in[10], *b2q = (const float*)d_in[11];
  const float *w2k = (const float*)d_in[12], *b2k = (const float*)d_in[13];
  const float *w2v = (const float*)d_in[14], *b2v = (const float*)d_in[15];
  const float *w2s = (const float*)d_in[16], *b2s = (const float*)d_in[17];
  float* out = (float*)d_out;

  // Workspace carve-up (~155 MB total); XBF region is reused as bf16 h after layer 1.
  char* p = (char*)d_ws;
  unsigned short* XBF = (unsigned short*)p; p += (size_t)N_NODES * 128 * 2;
  unsigned short* WT1 = (unsigned short*)p; p += (size_t)D_CAT * K_DIM * 2;
  unsigned short* WT2 = (unsigned short*)p; p += (size_t)D_CAT * K_DIM * 2;
  float* B1 = (float*)p; p += D_CAT * 4;
  float* B2 = (float*)p; p += D_CAT * 4;
  float* C  = (float*)p; p += (size_t)N_NODES * D_CAT * 4;   // q|k|v|s (reused per layer)
  float* LG = (float*)p; p += (size_t)N_EDGES * 4 * 4;       // logits (layer1 uses 4E)
  unsigned int* NMAX = (unsigned int*)p; p += (size_t)N_NODES * 4 * 4;
  float* NSUM = (float*)p; p += (size_t)N_NODES * 4 * 4;
  float* ACC  = (float*)p; p += (size_t)N_NODES * 128 * 4;
  unsigned short* HBF = XBF;

  const int NB_NC  = (N_NODES * 128 + 255) / 256;   // node*channel grids
  const int NB_NH  = (N_NODES * 4 + 255) / 256;
  const int NB_E   = N_EDGES / 8;                   // one wave per edge, 8 waves/block
  dim3 gemmGrid((N_NODES + 127) / 128, D_CAT / 128);

  // ---- prep
  k_cvt_bf16<<<NB_NC, 256, 0, stream>>>(x, XBF, N_NODES * 128);
  k_pack_w<<<(D_CAT * K_DIM + 255) / 256, 256, 0, stream>>>(
      w1q, w1k, w1v, w1s, b1q, b1k, b1v, b1s, WT1, B1);
  k_pack_w<<<(D_CAT * K_DIM + 255) / 256, 256, 0, stream>>>(
      w2q, w2k, w2v, w2s, b2q, b2k, b2v, b2s, WT2, B2);

  // ---- layer 1
  k_fill_u32<<<NB_NC, 256, 0, stream>>>((unsigned int*)ACC, 0u, N_NODES * 128);
  k_fill_u32<<<NB_NH, 256, 0, stream>>>((unsigned int*)NSUM, 0u, N_NODES * 4);
  k_fill_u32<<<NB_NH, 256, 0, stream>>>(NMAX, ENC_NEG_INF, N_NODES * 4);
  k_gemm_bf16<<<gemmGrid, 256, 0, stream>>>(XBF, WT1, B1, C, N_NODES);
  k_edge_logits4<<<NB_E, 256, 0, stream>>>(ei, C, LG, NMAX, N_EDGES);
  k_edge_agg4<<<NB_E, 256, 0, stream>>>(ei, C, LG, NMAX, NSUM, ACC, N_EDGES);
  k_final_l1<<<NB_NC, 256, 0, stream>>>(ACC, NSUM, C, HBF);

  // ---- layer 2 (heads=1)
  k_fill_u32<<<NB_NC, 256, 0, stream>>>((unsigned int*)ACC, 0u, N_NODES * 128);
  k_fill_u32<<<NB_NH, 256, 0, stream>>>((unsigned int*)NSUM, 0u, N_NODES);
  k_fill_u32<<<NB_NH, 256, 0, stream>>>(NMAX, ENC_NEG_INF, N_NODES);
  k_gemm_bf16<<<gemmGrid, 256, 0, stream>>>(HBF, WT2, B2, C, N_NODES);
  k_edge_logits1<<<NB_E, 256, 0, stream>>>(ei, C, LG, NMAX, N_EDGES);
  k_edge_agg1<<<NB_E, 256, 0, stream>>>(ei, C, LG, NMAX, NSUM, ACC, N_EDGES);
  k_final_l2<<<NB_NC, 256, 0, stream>>>(ACC, NSUM, C, out);
}
